// AlignmentNetwork_6399501271459
// MI455X (gfx1250) — matrix operand order repaired and checked by reading, not compiled, not run on hardware
//
#include <hip/hip_runtime.h>
#include <math.h>

// ---------------------------------------------------------------------------
// CDNA5 (gfx1250) AlignmentNetwork: conv stacks as bf16 WMMA GEMMs + fused
// squared-distance attention logits + row softmax.
// Round 2: halo-padded conv inputs (no predicated loads / exec juggling),
// 16x80 strips per wave (A-frag reused across 5 WMMAs).
// ---------------------------------------------------------------------------

typedef __attribute__((ext_vector_type(16))) __bf16 v16bf;
typedef __attribute__((ext_vector_type(8)))  float  v8f;

#define TEMPERATURE 0.0005f
#define NB 5   // 16x(16*NB) output strip per wave; 80 divides Ten=400, Tde=2000

// ---- bf16 <-> f32 helpers (bit-exact RNE, no dependence on __bf16 arith) ----
__device__ __forceinline__ __bf16 f32_to_bf16(float f) {
  union { float f; unsigned u; } a; a.f = f;
  unsigned r = a.u + 0x7FFFu + ((a.u >> 16) & 1u);
  union { unsigned short s; __bf16 b; } o; o.s = (unsigned short)(r >> 16);
  return o.b;
}
__device__ __forceinline__ float bf16_to_f32(__bf16 b) {
  union { unsigned short s; __bf16 b; } i; i.b = b;
  union { unsigned u; float f; } o; o.u = ((unsigned)i.s) << 16;
  return o.f;
}

union BF16x16 { v16bf v; uint4 q[2]; };

// A-matrix 16x32 bf16 frag (ISA 7.12.2): lane m = l&15; half = l>>4
// holds K = {half*8..+7} U {16+half*8..+7}  (two 16B loads)
__device__ __forceinline__ v16bf load_a_frag(const __bf16* row, int half) {
  BF16x16 r;
  r.q[0] = *(const uint4*)(row + half * 8);
  r.q[1] = *(const uint4*)(row + half * 8 + 16);
  return r.v;
}
// B-matrix 32x16 bf16 frag: lane n = l&15; half = l>>4
// holds K = {half*8..+7} U {16+half*8..+7}  (two 16B loads)
__device__ __forceinline__ v16bf load_b_frag(const __bf16* col, int khalf) {
  BF16x16 r;
  r.q[0] = *(const uint4*)(col + khalf * 8);
  r.q[1] = *(const uint4*)(col + khalf * 8 + 16);
  return r.v;
}

// ---------------------------------------------------------------------------
// Layout conversion kernels (fp32 -> bf16, transpose, K padding, T halo)
// ---------------------------------------------------------------------------
// out[b][rowoff + t][c<Cp] = (bf16) in[b][c][t]; halo rows stay memset-zero
__global__ void cvt_transpose(const float* __restrict__ in, __bf16* __restrict__ out,
                              int Bb, int C, int T, int Cp, int Trows, int rowoff) {
  size_t tid = (size_t)blockIdx.x * blockDim.x + threadIdx.x;
  size_t total = (size_t)Bb * T * Cp;
  if (tid >= total) return;
  int c = (int)(tid % Cp);
  size_t bt = tid / Cp;
  int t = (int)(bt % T);
  int b = (int)(bt / T);
  float v = (c < C) ? in[((size_t)b * C + c) * T + t] : 0.0f;
  out[((size_t)b * Trows + rowoff + t) * Cp + c] = f32_to_bf16(v);
}

// conv3 weights: out[tap][co][ci<Kp] = (bf16) in[co][ci][tap]
__global__ void cvt_w3(const float* __restrict__ in, __bf16* __restrict__ out,
                       int M, int Cin, int Kp) {
  size_t tid = (size_t)blockIdx.x * blockDim.x + threadIdx.x;
  size_t total = (size_t)3 * M * Kp;
  if (tid >= total) return;
  int ci = (int)(tid % Kp);
  size_t mo = tid / Kp;
  int co = (int)(mo % M);
  int tap = (int)(mo / M);
  float v = (ci < Cin) ? in[((size_t)co * Cin + ci) * 3 + tap] : 0.0f;
  out[tid] = f32_to_bf16(v);
}

// conv1 weights: out[co][ci<Kp] = (bf16) in[co][ci]
__global__ void cvt_w1(const float* __restrict__ in, __bf16* __restrict__ out,
                       int M, int Cin, int Kp) {
  size_t tid = (size_t)blockIdx.x * blockDim.x + threadIdx.x;
  size_t total = (size_t)M * Kp;
  if (tid >= total) return;
  int ci = (int)(tid % Kp);
  int co = (int)(tid / Kp);
  float v = (ci < Cin) ? in[(size_t)co * Cin + ci] : 0.0f;
  out[tid] = f32_to_bf16(v);
}

// ---------------------------------------------------------------------------
// Conv-as-GEMM, one wave per 16 x (16*NB) output strip.
//   W: [TAPS][M][Kp] bf16 (row m, K contiguous)              -> WMMA A
//   X: [Bb][Trows][Kp] bf16, t=0 at row `rowoff` (T halo)    -> WMMA B
//   Y: [Bb][T][Mstride] bf16, Y[b][t][co] = act(sum+bias[co])
// TAPS==3: k=3 pad=1 conv via 3 shifted accumulations; halo rows are zero,
// so all loads are unconditional and EXEC stays all-1s for WMMA.
// ---------------------------------------------------------------------------
template <int TAPS>
__global__ __launch_bounds__(32)
void gemm_tile(const __bf16* __restrict__ W, const __bf16* __restrict__ X,
               const float* __restrict__ bias, __bf16* __restrict__ Y,
               int M, int T, int Kp, int Trows, int rowoff, int Mstride, int relu) {
  const int l  = threadIdx.x;
  const int lm = l & 15, hs = l >> 4;
  const int mt = blockIdx.x, nt = blockIdx.y, b = blockIdx.z;
  const int m  = mt * 16 + lm;
  const int t0 = nt * (16 * NB);

  v8f acc[NB] = {};
#pragma unroll
  for (int tap = 0; tap < TAPS; ++tap) {
    const __bf16* Arow = W + ((size_t)tap * M + m) * Kp;
    const int tb = t0 + lm + tap - (TAPS >> 1) + rowoff;
    const __bf16* Bcol[NB];
#pragma unroll
    for (int s = 0; s < NB; ++s)
      Bcol[s] = X + ((size_t)b * Trows + tb + s * 16) * Kp;

    for (int k0 = 0; k0 < Kp; k0 += 32) {
      v16bf a = load_a_frag(Arow + k0, hs);   // reused by NB WMMAs
#pragma unroll
      for (int s = 0; s < NB; ++s) {
        v16bf bb = load_b_frag(Bcol[s] + k0, hs);
        acc[s] = __builtin_amdgcn_wmma_f32_16x16x32_bf16(
            false, a, false, bb, (short)0, acc[s], false, false);
      }
    }
  }

  // C/D layout: lane holds n = l&15, rows m = hs*8 + v -> 8 contiguous
  // channels per lane, stored as one 16B bf16 packet (next layer's B layout).
  const int mb = mt * 16 + hs * 8;
  float bv[8];
#pragma unroll
  for (int v = 0; v < 8; ++v) bv[v] = bias[mb + v];
#pragma unroll
  for (int s = 0; s < NB; ++s) {
    const int t = t0 + s * 16 + lm;
    __bf16 out8[8];
#pragma unroll
    for (int v = 0; v < 8; ++v) {
      float x = acc[s][v] + bv[v];
      if (relu) x = fmaxf(x, 0.0f);
      out8[v] = f32_to_bf16(x);
    }
    *(uint4*)(Y + ((size_t)b * T + t) * Mstride + mb) = *(const uint4*)out8;
  }
}

// |feat|^2 over real channels (same bf16 values WMMA consumes, so
// dist = |q~|^2 + |k~|^2 - 2 q~.k~ stays consistent / >= 0)
__global__ void sqnorm(const __bf16* __restrict__ f, float* __restrict__ out,
                       int n, int Kp, int C) {
  int tid = blockIdx.x * blockDim.x + threadIdx.x;
  if (tid >= n) return;
  const __bf16* p = f + (size_t)tid * Kp;
  float s = 0.0f;
  for (int c = 0; c < C; ++c) { float x = bf16_to_f32(p[c]); s += x * x; }
  out[tid] = s;
}

// ---------------------------------------------------------------------------
// logp[b][td][te] = mask ? -TEMP*(q2[td]+k2[te]-2*q.k) : -inf
// One wave per 16(td) x 80(te) strip; Kp=96 fixed -> fully unrolled 15 WMMAs,
// q frags reused 5x.
// ---------------------------------------------------------------------------
__global__ __launch_bounds__(32)
void attn_logp_kernel(const __bf16* __restrict__ qf, const __bf16* __restrict__ kf,
                      const float* __restrict__ q2, const float* __restrict__ k2,
                      const int* __restrict__ mask, float* __restrict__ logp,
                      int Tde, int Ten) {
  constexpr int KP = 96;
  const int l  = threadIdx.x;
  const int lm = l & 15, hs = l >> 4;
  const int mt = blockIdx.x, nt = blockIdx.y, b = blockIdx.z;

  const __bf16* qrow = qf + ((size_t)b * Tde + mt * 16 + lm) * KP;
  const __bf16* kcol[NB];
#pragma unroll
  for (int s = 0; s < NB; ++s)
    kcol[s] = kf + ((size_t)b * Ten + nt * (16 * NB) + s * 16 + lm) * KP;

  v8f acc[NB] = {};
#pragma unroll
  for (int k0 = 0; k0 < KP; k0 += 32) {
    v16bf a = load_a_frag(qrow + k0, hs);
#pragma unroll
    for (int s = 0; s < NB; ++s) {
      v16bf bb = load_b_frag(kcol[s] + k0, hs);
      acc[s] = __builtin_amdgcn_wmma_f32_16x16x32_bf16(
          false, a, false, bb, (short)0, acc[s], false, false);
    }
  }

  const int mbase = mt * 16 + hs * 8;
  float q2v[8];
#pragma unroll
  for (int v = 0; v < 8; ++v) q2v[v] = q2[b * Tde + mbase + v];
#pragma unroll
  for (int s = 0; s < NB; ++s) {
    const int te    = nt * (16 * NB) + s * 16 + lm;
    const float k2v = k2[b * Ten + te];
    const bool mk   = mask[b * Ten + te] != 0;
#pragma unroll
    for (int v = 0; v < 8; ++v) {
      float d  = q2v[v] + k2v - 2.0f * acc[s][v];
      float lp = mk ? (-TEMPERATURE * d) : -INFINITY;
      logp[((size_t)b * Tde + mbase + v) * Ten + te] = lp;
    }
  }
}

// Row softmax over Ten; 4 waves per block, one wave per (b,td) row.
// logp rows are L2-resident (102 MB < 192 MB global L2) between kernels.
__global__ __launch_bounds__(128)
void softmax_rows(const float* __restrict__ lp, float* __restrict__ attn,
                  int nrows, int Ten) {
  const int lane = threadIdx.x & 31;
  const int row  = blockIdx.x * 4 + (threadIdx.x >> 5);
  if (row >= nrows) return;
  const float* src = lp + (size_t)row * Ten;

  float m = -INFINITY;
  for (int i = lane; i < Ten; i += 32) m = fmaxf(m, src[i]);
  for (int o = 16; o > 0; o >>= 1) m = fmaxf(m, __shfl_xor(m, o, 32));

  float s = 0.0f;
  for (int i = lane; i < Ten; i += 32) s += __expf(src[i] - m);
  for (int o = 16; o > 0; o >>= 1) s += __shfl_xor(s, o, 32);

  const float inv = 1.0f / s;
  float* dst = attn + (size_t)row * Ten;
  for (int i = lane; i < Ten; i += 32) dst[i] = __expf(src[i] - m) * inv;
}

// ---------------------------------------------------------------------------
extern "C" void kernel_launch(void* const* d_in, const int* in_sizes, int n_in,
                              void* d_out, int out_size, void* d_ws, size_t ws_size,
                              hipStream_t stream) {
  (void)in_sizes; (void)n_in; (void)out_size; (void)ws_size;
  const float* queries = (const float*)d_in[0];   // [32,80,2000]
  const float* keys    = (const float*)d_in[1];   // [32,512,400]
  const int*   mask    = (const int*)d_in[2];     // [32,1,400]
  const float* kw1 = (const float*)d_in[3];  const float* kb1 = (const float*)d_in[4];
  const float* kw2 = (const float*)d_in[5];  const float* kb2 = (const float*)d_in[6];
  const float* qw1 = (const float*)d_in[7];  const float* qb1 = (const float*)d_in[8];
  const float* qw2 = (const float*)d_in[9];  const float* qb2 = (const float*)d_in[10];
  const float* qw3 = (const float*)d_in[11]; const float* qb3 = (const float*)d_in[12];

  constexpr int Bb = 32, Cq = 80, Ck = 512, Ca = 80, Tde = 2000, Ten = 400;
  constexpr int CkH = 1024, CqH = 160, P96 = 96;   // K padded 80 -> 96
  constexpr int HALO = 16;                          // zero rows on each T edge
  constexpr int TenH = Ten + 2 * HALO, TdeH = Tde + 2 * HALO;

  // ---- workspace layout (bf16 activations [b][t][c], 256B aligned)
  char* w = (char*)d_ws;
  size_t off = 0;
  auto alloc = [&](size_t bytes) -> void* {
    void* p = w + off; off = (off + bytes + 255) & ~(size_t)255; return p;
  };
  __bf16* Xkt   = (__bf16*)alloc((size_t)Bb * TenH * Ck  * 2);  // haloed
  __bf16* Xqt   = (__bf16*)alloc((size_t)Bb * TdeH * P96 * 2);  // haloed
  __bf16* kw1t  = (__bf16*)alloc((size_t)3 * CkH * Ck  * 2);
  __bf16* qw1t  = (__bf16*)alloc((size_t)3 * CqH * P96 * 2);
  __bf16* kw2b  = (__bf16*)alloc((size_t)Ca * CkH * 2);
  __bf16* qw2b  = (__bf16*)alloc((size_t)Cq * CqH * 2);
  __bf16* qw3b  = (__bf16*)alloc((size_t)Ca * P96 * 2);
  __bf16* Y1    = (__bf16*)alloc((size_t)Bb * Ten * CkH * 2);
  __bf16* Z1    = (__bf16*)alloc((size_t)Bb * Tde * CqH * 2);
  __bf16* Z2    = (__bf16*)alloc((size_t)Bb * Tde * P96 * 2);
  __bf16* qfeat = (__bf16*)alloc((size_t)Bb * Tde * P96 * 2);
  __bf16* kfeat = (__bf16*)alloc((size_t)Bb * Ten * P96 * 2);
  float*  q2    = (float*)alloc((size_t)Bb * Tde * 4);
  float*  k2    = (float*)alloc((size_t)Bb * Ten * 4);

  hipMemsetAsync(d_ws, 0, off, stream);  // zeros K-pad columns + T halos

  auto gr = [](size_t n) { return (unsigned)((n + 255) / 256); };

  // ---- precision/layout conversion
  cvt_transpose<<<gr((size_t)Bb * Ten * Ck),  256, 0, stream>>>(
      keys,    Xkt, Bb, Ck, Ten, Ck,  TenH, HALO);
  cvt_transpose<<<gr((size_t)Bb * Tde * P96), 256, 0, stream>>>(
      queries, Xqt, Bb, Cq, Tde, P96, TdeH, HALO);
  cvt_w3<<<gr((size_t)3 * CkH * Ck),  256, 0, stream>>>(kw1, kw1t, CkH, Ck, Ck);
  cvt_w3<<<gr((size_t)3 * CqH * P96), 256, 0, stream>>>(qw1, qw1t, CqH, Cq, P96);
  cvt_w1<<<gr((size_t)Ca * CkH), 256, 0, stream>>>(kw2, kw2b, Ca, CkH, CkH);
  cvt_w1<<<gr((size_t)Cq * CqH), 256, 0, stream>>>(qw2, qw2b, Cq, CqH, CqH);
  cvt_w1<<<gr((size_t)Ca * P96), 256, 0, stream>>>(qw3, qw3b, Ca, Cq, P96);

  // ---- key path: conv3(512->1024)+ReLU, conv1(1024->80)
  gemm_tile<3><<<dim3(CkH / 16, Ten / (16 * NB), Bb), 32, 0, stream>>>(
      kw1t, Xkt, kb1, Y1, CkH, Ten, Ck, TenH, HALO, CkH, 1);
  gemm_tile<1><<<dim3(Ca / 16, Ten / (16 * NB), Bb), 32, 0, stream>>>(
      kw2b, Y1, kb2, kfeat, Ca, Ten, CkH, Ten, 0, P96, 0);

  // ---- query path: conv3(80->160)+ReLU, conv1(160->80)+ReLU, conv1(80->80)
  gemm_tile<3><<<dim3(CqH / 16, Tde / (16 * NB), Bb), 32, 0, stream>>>(
      qw1t, Xqt, qb1, Z1, CqH, Tde, P96, TdeH, HALO, CqH, 1);
  gemm_tile<1><<<dim3(Cq / 16, Tde / (16 * NB), Bb), 32, 0, stream>>>(
      qw2b, Z1, qb2, Z2, Cq, Tde, CqH, Tde, 0, P96, 1);
  gemm_tile<1><<<dim3(Ca / 16, Tde / (16 * NB), Bb), 32, 0, stream>>>(
      qw3b, Z2, qb3, qfeat, Ca, Tde, P96, Tde, 0, P96, 0);

  // ---- squared norms (from the bf16 features WMMA consumes)
  sqnorm<<<gr((size_t)Bb * Tde), 256, 0, stream>>>(qfeat, q2, Bb * Tde, P96, Ca);
  sqnorm<<<gr((size_t)Bb * Ten), 256, 0, stream>>>(kfeat, k2, Bb * Ten, P96, Ca);

  // ---- outputs: [attn | logp], each [32,2000,400] fp32
  float* attn = (float*)d_out;
  float* logp = attn + (size_t)Bb * Tde * Ten;

  attn_logp_kernel<<<dim3(Tde / 16, Ten / (16 * NB), Bb), 32, 0, stream>>>(
      qfeat, kfeat, q2, k2, mask, logp, Tde, Ten);
  softmax_rows<<<(Bb * Tde) / 4, 128, 0, stream>>>(logp, attn, Bb * Tde, Ten);
}
